// GCN_54863912239236
// MI455X (gfx1250) — matrix operand order repaired and checked by reading, not compiled, not run on hardware
//
#include <hip/hip_runtime.h>

typedef __attribute__((ext_vector_type(16))) _Float16 v16h;
typedef __attribute__((ext_vector_type(8)))  float    v8f;

// ---------------------------------------------------------------------------
// Pack W[KDIM, NOUT] (f32 row-major) into per-wave WMMA B-fragments (f16).
// Fragment f = (ksIdx * (NOUT/16) + colTile); within a fragment, lane L holds
// 16 contiguous halves: col N = L%16, g = L/16, half i -> K = ks + g*16 + i.
// Memory layout: Bfrag[(f*32 + lane)*16 + i]  (32 B contiguous per lane).
// ---------------------------------------------------------------------------
template<int KDIM, int NOUT>
__global__ void pack_w_frag(const float* __restrict__ W,
                            _Float16* __restrict__ Bfrag) {
  constexpr int CT = NOUT / 16;
  const int idx = blockIdx.x * blockDim.x + threadIdx.x;   // one half-element
  if (idx >= KDIM * NOUT) return;
  const int i    = idx & 15;
  const int lane = (idx >> 4) & 31;
  const int f    = idx >> 9;
  const int ct     = f % CT;
  const int ksIdx  = f / CT;
  const int g = lane >> 4;
  const int n = lane & 15;
  const int kb = ksIdx * 32 + g * 16 + i;
  Bfrag[idx] = (_Float16)W[(size_t)kb * NOUT + ct * 16 + n];
}

// ---------------------------------------------------------------------------
// Dense GEMM with fused input activation:
//   out[N, NOUT] = act(A) @ W,  act(v) = BIASRELU ? relu(v + bias[k]) : v
// Packed B fragments (KDIM*NOUT*2 bytes) are staged once per workgroup into
// LDS via gfx1250 async global->LDS (global_load_async_to_lds_b128, drained
// with s_wait_asynccnt). One wave32 then computes a full 16 x NOUT row-stripe:
// A fragment built once per K-step (f32 load -> bias/relu -> cvt f16), reused
// by NOUT/16 back-to-back WMMAs whose B operands come from LDS.
// ---------------------------------------------------------------------------
template<int KDIM, int NOUT, bool BIASRELU>
__global__ void gemm_wmma_f16(const float* __restrict__ A,
                              const _Float16* __restrict__ Bfrag,
                              const float* __restrict__ bias,
                              float* __restrict__ out,
                              int nRows, int rowTiles) {
  constexpr int CT = NOUT / 16;
  __shared__ _Float16 ldsB[KDIM * NOUT];

  // ---- Stage packed weights into LDS (async copy, 16 B per lane-issue) ----
  {
    const unsigned ldsBase = (unsigned)(uintptr_t)(&ldsB[0]);
    constexpr int nChunks = (KDIM * NOUT) / 8;           // 16-byte chunks
    for (int c = threadIdx.x; c < nChunks; c += blockDim.x) {
      const unsigned laddr = ldsBase + (unsigned)c * 16u;
      const void* gaddr = (const char*)Bfrag + (size_t)c * 16u;
      asm volatile("global_load_async_to_lds_b128 %0, %1, off"
                   :: "v"(laddr), "v"(gaddr) : "memory");
    }
    asm volatile("s_wait_asynccnt 0x0" ::: "memory");
  }
  __syncthreads();

  const int lane = threadIdx.x & 31;
  int tileM = (blockIdx.x * blockDim.x + threadIdx.x) >> 5;
  if (tileM > rowTiles - 1) tileM = rowTiles - 1;  // clamp: keep barrier-safe,
                                                   // dup waves rewrite same data
  const int g = lane >> 4;
  const int m = lane & 15;
  int row = tileM * 16 + m;
  if (row > nRows - 1) row = nRows - 1;

  v8f acc[CT];
  #pragma unroll
  for (int ct = 0; ct < CT; ++ct) acc[ct] = (v8f){};

  #pragma unroll
  for (int ks = 0; ks < KDIM; ks += 32) {
    const int ksIdx = ks >> 5;
    const float* arow = A + (size_t)row * KDIM + ks + g * 16;

    float4 p0 = *reinterpret_cast<const float4*>(arow + 0);
    float4 p1 = *reinterpret_cast<const float4*>(arow + 4);
    float4 p2 = *reinterpret_cast<const float4*>(arow + 8);
    float4 p3 = *reinterpret_cast<const float4*>(arow + 12);

    if (BIASRELU) {
      const float* brow = bias + ks + g * 16;
      const float4 q0 = *reinterpret_cast<const float4*>(brow + 0);
      const float4 q1 = *reinterpret_cast<const float4*>(brow + 4);
      const float4 q2 = *reinterpret_cast<const float4*>(brow + 8);
      const float4 q3 = *reinterpret_cast<const float4*>(brow + 12);
      p0.x = fmaxf(p0.x + q0.x, 0.f); p0.y = fmaxf(p0.y + q0.y, 0.f);
      p0.z = fmaxf(p0.z + q0.z, 0.f); p0.w = fmaxf(p0.w + q0.w, 0.f);
      p1.x = fmaxf(p1.x + q1.x, 0.f); p1.y = fmaxf(p1.y + q1.y, 0.f);
      p1.z = fmaxf(p1.z + q1.z, 0.f); p1.w = fmaxf(p1.w + q1.w, 0.f);
      p2.x = fmaxf(p2.x + q2.x, 0.f); p2.y = fmaxf(p2.y + q2.y, 0.f);
      p2.z = fmaxf(p2.z + q2.z, 0.f); p2.w = fmaxf(p2.w + q2.w, 0.f);
      p3.x = fmaxf(p3.x + q3.x, 0.f); p3.y = fmaxf(p3.y + q3.y, 0.f);
      p3.z = fmaxf(p3.z + q3.z, 0.f); p3.w = fmaxf(p3.w + q3.w, 0.f);
    }

    v16h a;
    a[0]  = (_Float16)p0.x; a[1]  = (_Float16)p0.y;
    a[2]  = (_Float16)p0.z; a[3]  = (_Float16)p0.w;
    a[4]  = (_Float16)p1.x; a[5]  = (_Float16)p1.y;
    a[6]  = (_Float16)p1.z; a[7]  = (_Float16)p1.w;
    a[8]  = (_Float16)p2.x; a[9]  = (_Float16)p2.y;
    a[10] = (_Float16)p2.z; a[11] = (_Float16)p2.w;
    a[12] = (_Float16)p3.x; a[13] = (_Float16)p3.y;
    a[14] = (_Float16)p3.z; a[15] = (_Float16)p3.w;

    #pragma unroll
    for (int ct = 0; ct < CT; ++ct) {
      const v16h b = *reinterpret_cast<const v16h*>(
          &ldsB[((size_t)(ksIdx * CT + ct) * 32 + lane) * 16]);
      // (neg_a, A, neg_b, B, c_mod, C, reuse_a, reuse_b)
      acc[ct] = __builtin_amdgcn_wmma_f32_16x16x32_f16(false, a, false, b,
                                                       (short)0, acc[ct],
                                                       false, false);
    }
  }

  #pragma unroll
  for (int ct = 0; ct < CT; ++ct) {
    #pragma unroll
    for (int r = 0; r < 8; ++r) {
      const int mm = tileM * 16 + r + g * 8;
      if (mm < nRows)
        out[(size_t)mm * NOUT + ct * 16 + m] = acc[ct][r];
    }
  }
}

// ---------------------------------------------------------------------------
// SpMM scatter-add: agg[dst] += w * support[src]; one wave per edge,
// lane covers D/32 contiguous floats (128 -> float4/lane, 64 -> float2/lane).
// ---------------------------------------------------------------------------
template<int D>
__global__ void spmm_atomic(const float* __restrict__ support,
                            const int* __restrict__ esrc,
                            const int* __restrict__ edst,
                            const float* __restrict__ ew,
                            float* __restrict__ agg, int E) {
  const int lane = threadIdx.x & 31;
  const int wave = (blockIdx.x * blockDim.x + threadIdx.x) >> 5;
  const int nWaves = (gridDim.x * blockDim.x) >> 5;
  constexpr int EPL = D / 32;

  for (int e = wave; e < E; e += nWaves) {
    const int   s = esrc[e];
    const int   d = edst[e];
    const float w = ew[e];
    const float* sp = support + (size_t)s * D + lane * EPL;
    float*       ap = agg     + (size_t)d * D + lane * EPL;
    if constexpr (EPL == 4) {
      const float4 v = *reinterpret_cast<const float4*>(sp);
      atomicAdd(ap + 0, w * v.x);
      atomicAdd(ap + 1, w * v.y);
      atomicAdd(ap + 2, w * v.z);
      atomicAdd(ap + 3, w * v.w);
    } else {
      const float2 v = *reinterpret_cast<const float2*>(sp);
      atomicAdd(ap + 0, w * v.x);
      atomicAdd(ap + 1, w * v.y);
    }
  }
}

// ---------------------------------------------------------------------------
// Fused bias + log_softmax over 64 features; one wave32 per row,
// 2 elements per lane, shuffle-based max / sum-exp reductions (wave32 width).
// ---------------------------------------------------------------------------
__global__ void bias_logsoftmax64(const float* __restrict__ agg,
                                  const float* __restrict__ bias,
                                  float* __restrict__ out, int nRows) {
  const int lane = threadIdx.x & 31;
  const int wave = (blockIdx.x * blockDim.x + threadIdx.x) >> 5;
  if (wave >= nRows) return;

  const float* r = agg + (size_t)wave * 64;
  const float v0 = r[lane]      + bias[lane];
  const float v1 = r[lane + 32] + bias[lane + 32];

  float m = fmaxf(v0, v1);
  #pragma unroll
  for (int o = 16; o > 0; o >>= 1) m = fmaxf(m, __shfl_xor(m, o, 32));

  float s = expf(v0 - m) + expf(v1 - m);
  #pragma unroll
  for (int o = 16; o > 0; o >>= 1) s += __shfl_xor(s, o, 32);

  const float lse = m + logf(s);
  float* o0 = out + (size_t)wave * 64;
  o0[lane]      = v0 - lse;
  o0[lane + 32] = v1 - lse;
}

// ---------------------------------------------------------------------------
extern "C" void kernel_launch(void* const* d_in, const int* in_sizes, int n_in,
                              void* d_out, int out_size, void* d_ws, size_t ws_size,
                              hipStream_t stream) {
  const float* x    = (const float*)d_in[0];
  const int*   esrc = (const int*)  d_in[1];
  const int*   edst = (const int*)  d_in[2];
  const float* ew   = (const float*)d_in[3];
  const float* W1   = (const float*)d_in[4];
  const float* b1   = (const float*)d_in[5];
  const float* W2   = (const float*)d_in[6];
  const float* b2   = (const float*)d_in[7];
  const float* W3   = (const float*)d_in[8];
  const float* b3   = (const float*)d_in[9];
  float* out = (float*)d_out;

  const int N = in_sizes[0] / 128;   // 50000
  const int E = in_sizes[1];         // 800000

  char* ws = (char*)d_ws;
  float* t    = (float*)ws;                              // support   N*128 f32
  float* agg  = (float*)(ws + (size_t)N * 128 * 4);      // aggregate N*128 f32
  char*  wsW  = ws + (size_t)2 * N * 128 * 4;            // packed weights (f16)
  _Float16* Bf1 = (_Float16*)(wsW);                      // 128*128 halves
  _Float16* Bf2 = (_Float16*)(wsW + 128 * 128 * 2);      // 128*128 halves
  _Float16* Bf3 = (_Float16*)(wsW + 2 * 128 * 128 * 2);  // 128*64  halves

  const int rowTiles = (N + 15) / 16;
  const dim3 blk(256);
  const int gemmBlocks = (rowTiles + 7) / 8;             // 8 waves / block
  const int spmmBlocks = 2048;                           // grid-stride edges
  const int lsBlocks = (N + 7) / 8;

  // ---- Pack weights into WMMA B-fragment layout (tiny, once per call) ----
  pack_w_frag<128, 128><<<(128 * 128 + 255) / 256, blk, 0, stream>>>(W1, Bf1);
  pack_w_frag<128, 128><<<(128 * 128 + 255) / 256, blk, 0, stream>>>(W2, Bf2);
  pack_w_frag<128, 64><<<(128 * 64 + 255) / 256, blk, 0, stream>>>(W3, Bf3);

  // ---- Layer 1: agg = spmm(x @ W1) ----
  gemm_wmma_f16<128, 128, false><<<gemmBlocks, blk, 0, stream>>>(
      x, Bf1, nullptr, t, N, rowTiles);
  hipMemsetAsync(agg, 0, (size_t)N * 128 * 4, stream);
  spmm_atomic<128><<<spmmBlocks, blk, 0, stream>>>(t, esrc, edst, ew, agg, E);

  // ---- Layer 2: agg = spmm(relu(agg + b1) @ W2)  (bias+relu fused in A-load)
  gemm_wmma_f16<128, 128, true><<<gemmBlocks, blk, 0, stream>>>(
      agg, Bf2, b1, t, N, rowTiles);
  hipMemsetAsync(agg, 0, (size_t)N * 128 * 4, stream);
  spmm_atomic<128><<<spmmBlocks, blk, 0, stream>>>(t, esrc, edst, ew, agg, E);

  // ---- Layer 3: out = log_softmax(spmm(relu(agg + b2) @ W3) + b3) ----
  gemm_wmma_f16<128, 64, true><<<gemmBlocks, blk, 0, stream>>>(
      agg, Bf3, b2, t, N, rowTiles);
  hipMemsetAsync(agg, 0, (size_t)N * 64 * 4, stream);
  spmm_atomic<64><<<spmmBlocks, blk, 0, stream>>>(t, esrc, edst, ew, agg, E);
  bias_logsoftmax64<<<lsBlocks, blk, 0, stream>>>(agg, b3, out, N);
}
